// BasicBlock_18047452578754
// MI455X (gfx1250) — matrix operand order repaired and checked by reading, not compiled, not run on hardware
//
#include <hip/hip_runtime.h>
#include <hip/hip_bf16.h>

typedef _Float16 h16;
typedef __attribute__((ext_vector_type(16))) _Float16 v16h;
typedef __attribute__((ext_vector_type(8)))  _Float16 v8h;
typedef __attribute__((ext_vector_type(8)))  float    v8f;

#define BB  64     // batch rows per block
#define HS  136    // padded h-row stride in halves (bank-spread, 16B aligned)
#define FOFF (8192*24)

// sigmoid(x) = rcp(1 + exp(-x)) using hardware v_rcp_f32 (TRANS, co-executes with WMMA)
__device__ __forceinline__ float sigm(float x) {
    return __builtin_amdgcn_rcpf(1.0f + __expf(-x));
}
// tanh(x) = 1 - 2*rcp(exp(2x) + 1)
__device__ __forceinline__ float tanh_f(float x) {
    return 1.0f - 2.0f * __builtin_amdgcn_rcpf(__expf(2.0f * x) + 1.0f);
}

__global__ __launch_bounds__(256)
void tcn_gru_fused(const float* __restrict__ X,
                   const float* __restrict__ w1, const float* __restrict__ b1,
                   const float* __restrict__ w2, const float* __restrict__ b2,
                   const float* __restrict__ wd, const float* __restrict__ bd,
                   const float* __restrict__ f_wih, const float* __restrict__ f_whh,
                   const float* __restrict__ f_bih, const float* __restrict__ f_bhh,
                   const float* __restrict__ f_wout, const float* __restrict__ f_bout,
                   const float* __restrict__ e_wih, const float* __restrict__ e_whh,
                   const float* __restrict__ e_bih, const float* __restrict__ e_bhh,
                   const float* __restrict__ e_wout, const float* __restrict__ e_bout,
                   float* __restrict__ out)
{
    // LDS overlay: encoder phase uses xs(10240B)+y1(49152B); decoder reuses as
    // hbuf0(17408) hbuf1(17408) xcur(256) oacc(256).
    __shared__ __align__(16) char smem[59392];
    float* xs   = (float*)smem;                     // [64][5][8]
    h16*   y1   = (h16*)(smem + 10240);             // [64][3][128]
    h16*   hb0  = (h16*)smem;                       // [64][HS]
    h16*   hb1  = (h16*)(smem + 17408);             // [64][HS]
    float* xcur = (float*)(smem + 34816);           // [64]
    float* oacc = (float*)(smem + 35072);           // [64]

    const int tid = threadIdx.x;
    const int b0  = blockIdx.x * BB;
    const int est = blockIdx.y;                     // 0=forecast, 1=estimate

    // ---------------- encoder: stage x[b, t=91..95, f] ----------------
    for (int i = tid; i < BB * 5 * 8; i += 256) {
        int row = i / 40, rem = i % 40, t = rem / 8, f = rem % 8;
        xs[i] = X[((size_t)(b0 + row) * 96 + 91 + t) * 8 + f];
    }
    __syncthreads();

    // conv1 (causal, only t=93..95 needed): y1[row][t][h]
    for (int i = tid; i < BB * 3 * 128; i += 256) {
        int hh = i & 127, t = (i >> 7) % 3, row = i / 384;
        float a = b1[hh];
        #pragma unroll
        for (int k = 0; k < 3; ++k)
            #pragma unroll
            for (int f = 0; f < 8; ++f)
                a = fmaf(w1[hh * 24 + f * 3 + k], xs[(row * 5 + t + k) * 8 + f], a);
        y1[i] = (h16)fmaxf(a, 0.0f);
    }
    __syncthreads();

    // conv2 (last timestep only) + 1x1 residual + relu -> h0
    {
        const int hh = tid & 127;
        const int r0 = (tid >> 7) * 32;
        float acc[32];
        const float bb2 = b2[hh];
        #pragma unroll
        for (int r = 0; r < 32; ++r) acc[r] = bb2;
        for (int t = 0; t < 3; ++t)
            for (int ho = 0; ho < 16; ++ho) {
                float w[8];
                #pragma unroll
                for (int j = 0; j < 8; ++j) w[j] = w2[hh * 384 + (ho * 8 + j) * 3 + t];
                #pragma unroll
                for (int r = 0; r < 32; ++r) {
                    v8h y = *(const v8h*)&y1[((r0 + r) * 3 + t) * 128 + ho * 8];
                    #pragma unroll
                    for (int j = 0; j < 8; ++j)
                        acc[r] = fmaf((float)y[j], w[j], acc[r]);
                }
            }
        float hreg[32];
        const float bdd = bd[hh];
        #pragma unroll
        for (int r = 0; r < 32; ++r) {
            float res = bdd;
            #pragma unroll
            for (int f = 0; f < 8; ++f)
                res = fmaf(wd[hh * 8 + f], xs[((r0 + r) * 5 + 4) * 8 + f], res);
            hreg[r] = fmaxf(fmaxf(acc[r], 0.0f) + res, 0.0f);
        }
        __syncthreads();   // done with xs/y1 -> reuse LDS as h buffers
        #pragma unroll
        for (int r = 0; r < 32; ++r)
            hb0[(r0 + r) * HS + hh] = (h16)hreg[r];
    }

    // ---------------- GRU decoder ----------------
    const float* Wih  = est ? e_wih  : f_wih;
    const float* Whh  = est ? e_whh  : f_whh;
    const float* Bih  = est ? e_bih  : f_bih;
    const float* Bhh  = est ? e_bhh  : f_bhh;
    const float* Wout = est ? e_wout : f_wout;
    const float  bo   = est ? e_bout[0] : f_bout[0];
    const int    nst  = est ? 96 : 24;

    if (tid < BB) {
        size_t b = (size_t)(b0 + tid);
        xcur[tid] = est ? X[b * 96 * 8] : X[(b * 96 + 95) * 8];
        oacc[tid] = 0.0f;
    }

    const int wv = tid >> 5;           // wave id 0..7 -> h-columns [16wv,16wv+16)
    const int ln = tid & 15;           // N column / M row within tile
    const int hf = (tid >> 4) & 1;     // lane half
    const int col = wv * 16 + ln;

    float wihg[3], bihg[3], bhhg[3];
    #pragma unroll
    for (int g = 0; g < 3; ++g) {
        int j = g * 128 + col;
        wihg[g] = Wih[j]; bihg[g] = Bih[j]; bhhg[g] = Bhh[j];
    }
    const float woutc = Wout[col];

    // whh B-fragments resident in VGPRs: B[k,n] = whh[g*128+16wv+n, k]
    // lanes 0-15: K = 32kc+0..15 ; lanes 16-31: K = 32kc+16..31 (contiguous k order used here, same as A)
    v16h Bw[3][4];
    #pragma unroll
    for (int g = 0; g < 3; ++g)
        #pragma unroll
        for (int kc = 0; kc < 4; ++kc) {
            const float* p = Whh + (size_t)(g * 128 + col) * 128 + kc * 32 + hf * 16;
            v16h t;
            #pragma unroll
            for (int q = 0; q < 16; ++q) t[q] = (h16)p[q];
            Bw[g][kc] = t;
        }
    __syncthreads();

    #pragma unroll 1
    for (int s = 0; s < nst; ++s) {
        const h16* hp = (s & 1) ? hb1 : hb0;
        h16*       hn = (s & 1) ? hb0 : hb1;

        #pragma unroll 1
        for (int m = 0; m < 4; ++m) {
            v8f acc[3];
            #pragma unroll
            for (int g = 0; g < 3; ++g) {
                v8f c;
                #pragma unroll
                for (int q = 0; q < 8; ++q) c[q] = bhhg[g];   // fold bhh into C
                acc[g] = c;
            }
            // gh[m-tile, 3 gate tiles] = h @ whh^T  via 12 WMMAs
            #pragma unroll
            for (int kc = 0; kc < 4; ++kc) {
                // A layout: lane holds M=ln; halves K = 16hf..16hf+15
                const h16* ap = hp + (m * 16 + ln) * HS + kc * 32 + hf * 16;
                v8h lo = *(const v8h*)ap;
                v8h hi = *(const v8h*)(ap + 8);
                v16h A = __builtin_shufflevector(lo, hi,
                          0,1,2,3,4,5,6,7,8,9,10,11,12,13,14,15);
                #pragma unroll
                for (int g = 0; g < 3; ++g)
                    acc[g] = __builtin_amdgcn_wmma_f32_16x16x32_f16(
                        false, A, false, Bw[g][kc], (short)0, acc[g], false, false);
            }
            // GRU elementwise in C layout: element v -> row = 16m + 8hf + v, col
            #pragma unroll
            for (int v = 0; v < 8; ++v) {
                int row = m * 16 + hf * 8 + v;
                float x    = xcur[row];
                float hold = (float)hp[row * HS + col];
                float r = sigm(acc[0][v] + fmaf(x, wihg[0], bihg[0]));
                float z = sigm(acc[1][v] + fmaf(x, wihg[1], bihg[1]));
                float n = tanh_f(fmaf(x, wihg[2], bihg[2]) + r * acc[2][v]);
                float hv = (1.0f - z) * n + z * hold;
                hn[row * HS + col] = (h16)hv;
                // out partial: reduce across the 16 N-lanes, then one ds_add
                float part = hv * woutc;
                part += __shfl_xor(part, 1, 32);
                part += __shfl_xor(part, 2, 32);
                part += __shfl_xor(part, 4, 32);
                part += __shfl_xor(part, 8, 32);
                if (ln == 0) atomicAdd(&oacc[row], part);
            }
        }
        __syncthreads();
        if (tid < BB) {
            size_t b = (size_t)(b0 + tid);
            float o = oacc[tid] + bo;
            if (est) out[FOFF + b * 96 + s] = X[(b * 96 + s) * 8] - o;
            else     out[b * 24 + s] = o;
            xcur[tid] = o;        // autoregressive feedback
            oacc[tid] = 0.0f;
        }
        __syncthreads();
    }
}

extern "C" void kernel_launch(void* const* d_in, const int* in_sizes, int n_in,
                              void* d_out, int out_size, void* d_ws, size_t ws_size,
                              hipStream_t stream) {
    (void)in_sizes; (void)n_in; (void)out_size; (void)d_ws; (void)ws_size;
    const float* X      = (const float*)d_in[0];
    const float* w1     = (const float*)d_in[1];
    const float* b1     = (const float*)d_in[2];
    const float* w2     = (const float*)d_in[3];
    const float* b2     = (const float*)d_in[4];
    const float* wd     = (const float*)d_in[5];
    const float* bd     = (const float*)d_in[6];
    const float* f_wih  = (const float*)d_in[7];
    const float* f_whh  = (const float*)d_in[8];
    const float* f_bih  = (const float*)d_in[9];
    const float* f_bhh  = (const float*)d_in[10];
    const float* f_wout = (const float*)d_in[11];
    const float* f_bout = (const float*)d_in[12];
    const float* e_wih  = (const float*)d_in[13];
    const float* e_whh  = (const float*)d_in[14];
    const float* e_bih  = (const float*)d_in[15];
    const float* e_bhh  = (const float*)d_in[16];
    const float* e_wout = (const float*)d_in[17];
    const float* e_bout = (const float*)d_in[18];
    float* outp = (float*)d_out;

    dim3 grid(8192 / BB, 2, 1);
    dim3 block(256, 1, 1);
    tcn_gru_fused<<<grid, block, 0, stream>>>(
        X, w1, b1, w2, b2, wd, bd,
        f_wih, f_whh, f_bih, f_bhh, f_wout, f_bout,
        e_wih, e_whh, e_bih, e_bhh, e_wout, e_bout,
        outp);
}
